// CrossAttention_48808008351766
// MI455X (gfx1250) — matrix operand order repaired and checked by reading, not compiled, not run on hardware
//
#include <hip/hip_runtime.h>
#include <hip/hip_bf16.h>

// ---------------------------------------------------------------------------
// Cross-attention block for MI455X (gfx1250), wave32, WMMA f16 16x16x32 path.
//
// dims: B=8, N=M=1024, Q_DIM=1024, K_DIM=768, H=16, D=64, SCALE=1/8
// outputs: out [B,N,1024] fp32  ||  attn.mean(heads) [B,N,1024] fp32
// ---------------------------------------------------------------------------

typedef __attribute__((ext_vector_type(16))) _Float16 v16h;
typedef __attribute__((ext_vector_type(8)))  _Float16 v8h;
typedef __attribute__((ext_vector_type(8)))  float    v8f;

#define B_SZ    8
#define N_SZ    1024
#define M_SZ    1024
#define QD      1024
#define KD      768
#define NH      16
#define HD      64
#define SCALE_F 0.125f

// ---------------------------------------------------------------------------
// fp32 -> f16 conversion (grid-stride)
// ---------------------------------------------------------------------------
__global__ __launch_bounds__(256)
void cvt_f32_to_f16(const float* __restrict__ src, _Float16* __restrict__ dst, int n) {
    int i = blockIdx.x * blockDim.x + threadIdx.x;
    int stride = gridDim.x * blockDim.x;
    for (; i < n; i += stride) dst[i] = (_Float16)src[i];
}

// ---------------------------------------------------------------------------
// WMMA GEMM (no LDS, barrier-free inner loop):
//   Y[r,c] = sum_k X[r,k] * W[c,k] + bias[c]        (X @ W^T + b)
//   X: [R,K] f16 row-major, W: [C,K] f16 row-major (torch Linear layout)
//   MODE 0: store f16 row-major Y[r*C+c]
//   MODE 1: store f16 transposed-per-head: vt[((b*NH+h)*HD+d)*M + m]
//           where r = b*1024+m, c = h*64+d          (for the V projection)
//   MODE 2: store f32 row-major (final output projection)
//
// block: 128 thr = 4 waves; block tile 64x64; wave: 16 rows x 64 cols.
// Fragments are loaded straight from global: A = two 16B chunks per lane,
// B = one 32B-aligned 32-byte chunk per lane (W rows are 32B aligned since
// K*2 in {1536, 2048}).  Redundant W reads hit the 192MB L2.
// ---------------------------------------------------------------------------
template<int MODE>
__global__ __launch_bounds__(128)
void gemm_wmma(const _Float16* __restrict__ X, const _Float16* __restrict__ W,
               const float* __restrict__ bias, void* __restrict__ out,
               int R, int K, int C) {
    const int t    = threadIdx.x;
    const int wave = t >> 5;
    const int lane = t & 31;
    const int ln   = lane & 15;
    const int hi   = lane >> 4;          // 0: lanes 0-15, 1: lanes 16-31
    const int c0   = blockIdx.x * 64;
    const int r0   = blockIdx.y * 64;

    // per-lane base pointers (K-invariant parts)
    const _Float16* pa = X + (size_t)(r0 + wave * 16 + ln) * K + (hi ? 16 : 0);
    const _Float16* pb0 = W + (size_t)(c0 +  0 + ln) * K + (hi ? 16 : 0);
    const _Float16* pb1 = W + (size_t)(c0 + 16 + ln) * K + (hi ? 16 : 0);
    const _Float16* pb2 = W + (size_t)(c0 + 32 + ln) * K + (hi ? 16 : 0);
    const _Float16* pb3 = W + (size_t)(c0 + 48 + ln) * K + (hi ? 16 : 0);

    v8f acc[4] = {{}, {}, {}, {}};

    #pragma unroll 2
    for (int k0 = 0; k0 < K; k0 += 32) {
        // A fragment: 16x32 f16, rows = this wave's 16 output rows
        v8h  alo = *(const v8h*)(pa + k0);
        v8h  ahi = *(const v8h*)(pa + k0 + 8);
        v16h a   = __builtin_shufflevector(alo, ahi,
                       0, 1, 2, 3, 4, 5, 6, 7, 8, 9, 10, 11, 12, 13, 14, 15);
        // B fragments: 32x16 f16 per column tile, B[k][n] = W[c][k]
        v16h b0 = *(const v16h*)(pb0 + k0);
        v16h b1 = *(const v16h*)(pb1 + k0);
        v16h b2 = *(const v16h*)(pb2 + k0);
        v16h b3 = *(const v16h*)(pb3 + k0);
        acc[0] = __builtin_amdgcn_wmma_f32_16x16x32_f16(false, a, false, b0, (short)0, acc[0], false, false);
        acc[1] = __builtin_amdgcn_wmma_f32_16x16x32_f16(false, a, false, b1, (short)0, acc[1], false, false);
        acc[2] = __builtin_amdgcn_wmma_f32_16x16x32_f16(false, a, false, b2, (short)0, acc[2], false, false);
        acc[3] = __builtin_amdgcn_wmma_f32_16x16x32_f16(false, a, false, b3, (short)0, acc[3], false, false);
    }

    // epilogue: bias + store.  D layout: lane col = ln, VGPR v -> row = v + 8*hi.
    #pragma unroll
    for (int ct = 0; ct < 4; ++ct) {
        int   c_g = c0 + ct * 16 + ln;
        float bv  = bias[c_g];
        #pragma unroll
        for (int v = 0; v < 8; ++v) {
            int   row = v + hi * 8;
            int   r_g = r0 + wave * 16 + row;
            float val = acc[ct][v] + bv;
            if (MODE == 0) {
                ((_Float16*)out)[(size_t)r_g * C + c_g] = (_Float16)val;
            } else if (MODE == 1) {
                int b = r_g >> 10, m = r_g & 1023;
                int h = c_g >> 6,  d = c_g & 63;
                ((_Float16*)out)[(size_t)((b * NH + h) * HD + d) * M_SZ + m] = (_Float16)val;
            } else {
                ((float*)out)[(size_t)r_g * C + c_g] = val;
            }
        }
    }
}

// ---------------------------------------------------------------------------
// Attention: one block per (batch, 16-row query tile), loops over 16 heads.
//   scores = (Q_h K_h^T) * SCALE ; P = softmax ; ctx = P V_h ; mean += P/16
// 128 threads = 4 waves.  Dynamic LDS:
//   sc  [16][1024] fp32  (scores -> probabilities, reused per head)
//   mn  [16][1024] fp32  (head-mean accumulator)
//   red [16][8] + rowv[16]
// ---------------------------------------------------------------------------
__global__ __launch_bounds__(128)
void attn_wmma(const _Float16* __restrict__ qh, const _Float16* __restrict__ kh,
               const _Float16* __restrict__ vt, _Float16* __restrict__ ctx,
               float* __restrict__ meanOut) {
    extern __shared__ float smem[];
    float* sc   = smem;                  // 16*1024
    float* mn   = smem + 16 * 1024;      // 16*1024
    float* red  = smem + 32 * 1024;      // 16*8
    float* rowv = red + 128;             // 16

    const int t    = threadIdx.x;
    const int wave = t >> 5;
    const int lane = t & 31;
    const int ln   = lane & 15;
    const int hi   = lane >> 4;
    const int b    = blockIdx.x >> 6;    // 64 row-tiles per batch
    const int n0   = (blockIdx.x & 63) * 16;

    for (int i = t; i < 16 * 1024; i += 128) mn[i] = 0.f;
    __syncthreads();

    for (int h = 0; h < NH; ++h) {
        // ---- scores: wave covers key columns [wave*256, wave*256+256) ----
        v16h aq[2];                       // Q tile 16x64 as two 16x32 fragments
        #pragma unroll
        for (int ks = 0; ks < 2; ++ks) {
            const _Float16* p = qh + (size_t)(b * N_SZ + n0 + ln) * QD
                                   + h * HD + ks * 32 + (hi ? 16 : 0);
            #pragma unroll
            for (int j = 0; j < 8; ++j) { aq[ks][j] = p[j]; aq[ks][8 + j] = p[8 + j]; }
        }
        for (int i = 0; i < 16; ++i) {
            int m0 = wave * 256 + i * 16;
            v8f accS = {};
            #pragma unroll
            for (int ks = 0; ks < 2; ++ks) {
                v16h bk;                  // B[k][n] = K[m0+n][k]
                const _Float16* p = kh + (size_t)(b * M_SZ + m0 + ln) * QD
                                       + h * HD + ks * 32 + (hi ? 16 : 0);
                #pragma unroll
                for (int j = 0; j < 16; ++j) bk[j] = p[j];
                accS = __builtin_amdgcn_wmma_f32_16x16x32_f16(
                    false, aq[ks], false, bk, (short)0, accS, false, false);
            }
            #pragma unroll
            for (int v = 0; v < 8; ++v)
                sc[(v + hi * 8) * 1024 + m0 + ln] = accS[v];
        }
        __syncthreads();

        // ---- softmax: row r handled by 8 threads, 128 cols each ----
        {
            const int r = t >> 3, seg = t & 7;
            float* srow = sc + r * 1024 + seg * 128;
            float mx = -3.4e38f;
            for (int c = 0; c < 128; ++c) {
                float s = srow[c] * SCALE_F;
                mx = fmaxf(mx, s);
            }
            red[r * 8 + seg] = mx;
            __syncthreads();
            if (t < 16) {
                float m2 = red[t * 8];
                for (int s2 = 1; s2 < 8; ++s2) m2 = fmaxf(m2, red[t * 8 + s2]);
                rowv[t] = m2;
            }
            __syncthreads();
            float rm  = rowv[r];
            float sum = 0.f;
            for (int c = 0; c < 128; ++c) {
                float e = __expf(srow[c] * SCALE_F - rm);
                srow[c] = e;
                sum += e;
            }
            red[r * 8 + seg] = sum;
            __syncthreads();
            if (t < 16) {
                float s2 = 0.f;
                for (int k = 0; k < 8; ++k) s2 += red[t * 8 + k];
                rowv[t] = 1.0f / s2;
            }
            __syncthreads();
            float inv = rowv[r];
            for (int c = 0; c < 128; ++c) {
                float p = srow[c] * inv;
                srow[c] = p;
                mn[r * 1024 + seg * 128 + c] += p;
            }
        }
        __syncthreads();

        // ---- ctx tile = P (16x1024) @ V_h (1024x64); wave owns d-tile ----
        {
            int d0 = wave * 16;
            v8f accO = {};
            for (int ms = 0; ms < 32; ++ms) {
                int  m = ms * 32;
                v16h ap;                  // A = P fragment (convert fp32->f16)
                {
                    const float* p = sc + ln * 1024 + m + (hi ? 16 : 0);
                    #pragma unroll
                    for (int j = 0; j < 8; ++j) {
                        ap[j]     = (_Float16)p[j];
                        ap[8 + j] = (_Float16)p[8 + j];
                    }
                }
                v16h bv;                  // B[k][n] = V[m][d0+n], from vt[b,h,d,m]
                {
                    const _Float16* p = vt + (size_t)((b * NH + h) * HD + d0 + ln) * M_SZ
                                           + m + (hi ? 16 : 0);
                    #pragma unroll
                    for (int j = 0; j < 16; ++j) bv[j] = p[j];
                }
                accO = __builtin_amdgcn_wmma_f32_16x16x32_f16(
                    false, ap, false, bv, (short)0, accO, false, false);
            }
            #pragma unroll
            for (int v = 0; v < 8; ++v) {
                int row = v + hi * 8;
                ctx[(size_t)(b * N_SZ + n0 + row) * QD + h * HD + d0 + ln] = (_Float16)accO[v];
            }
        }
        __syncthreads();
    }

    // ---- write head-mean (x 1/16) ----
    for (int i = t; i < 16 * 1024; i += 128) {
        int r = i >> 10, m = i & 1023;
        meanOut[(size_t)(b * N_SZ + n0 + r) * M_SZ + m] = mn[i] * 0.0625f;
    }
}

// ---------------------------------------------------------------------------
// Host launcher
// ---------------------------------------------------------------------------
extern "C" void kernel_launch(void* const* d_in, const int* in_sizes, int n_in,
                              void* d_out, int out_size, void* d_ws, size_t ws_size,
                              hipStream_t stream) {
    const float* q_f   = (const float*)d_in[0];
    const float* k_f   = (const float*)d_in[1];
    const float* v_f   = (const float*)d_in[2];
    const float* w_q   = (const float*)d_in[3];
    const float* b_q   = (const float*)d_in[4];
    const float* w_k   = (const float*)d_in[5];
    const float* b_k   = (const float*)d_in[6];
    const float* w_v   = (const float*)d_in[7];
    const float* b_v   = (const float*)d_in[8];
    const float* w_o   = (const float*)d_in[9];
    const float* b_o   = (const float*)d_in[10];
    float*       out   = (float*)d_out;

    const int R   = B_SZ * N_SZ;                       // 8192 rows
    const int nXQ = B_SZ * N_SZ * QD;                  // 8388608
    const int nXK = B_SZ * M_SZ * KD;                  // 6291456
    const int nWQ = QD * QD;                           // 1048576
    const int nWK = QD * KD;                           // 786432

    // workspace carve-up (f16 buffers, all offsets multiple of 256B)
    char* ws = (char*)d_ws;
    size_t off = 0;
    _Float16* xq16 = (_Float16*)(ws + off); off += (size_t)nXQ * 2;   // also reused as ctx
    _Float16* xk16 = (_Float16*)(ws + off); off += (size_t)nXK * 2;
    _Float16* xv16 = (_Float16*)(ws + off); off += (size_t)nXK * 2;
    _Float16* wq16 = (_Float16*)(ws + off); off += (size_t)nWQ * 2;
    _Float16* wk16 = (_Float16*)(ws + off); off += (size_t)nWK * 2;
    _Float16* wv16 = (_Float16*)(ws + off); off += (size_t)nWK * 2;
    _Float16* wo16 = (_Float16*)(ws + off); off += (size_t)nWQ * 2;
    _Float16* q_h  = (_Float16*)(ws + off); off += (size_t)nXQ * 2;
    _Float16* k_h  = (_Float16*)(ws + off); off += (size_t)nXQ * 2;
    _Float16* v_t  = (_Float16*)(ws + off); off += (size_t)nXQ * 2;
    _Float16* ctx  = xq16;   // alias: xq16 fully consumed before attention runs

    // 1) fp32 -> f16 conversions
    const int CT = 256;
    auto cvt = [&](const float* s, _Float16* d, int n) {
        int blocks = (n + CT * 8 - 1) / (CT * 8);
        cvt_f32_to_f16<<<blocks, CT, 0, stream>>>(s, d, n);
    };
    cvt(q_f, xq16, nXQ);
    cvt(k_f, xk16, nXK);
    cvt(v_f, xv16, nXK);
    cvt(w_q, wq16, nWQ);
    cvt(w_k, wk16, nWK);
    cvt(w_v, wv16, nWK);
    cvt(w_o, wo16, nWQ);

    // 2) projections (X @ W^T + b)
    dim3 gGemm(QD / 64, R / 64);   // 16 x 128
    gemm_wmma<0><<<gGemm, 128, 0, stream>>>(xq16, wq16, b_q, (void*)q_h, R, QD, QD);
    gemm_wmma<0><<<gGemm, 128, 0, stream>>>(xk16, wk16, b_k, (void*)k_h, R, KD, QD);
    gemm_wmma<1><<<gGemm, 128, 0, stream>>>(xv16, wv16, b_v, (void*)v_t, R, KD, QD);

    // 3) attention (scores + softmax + PV + head-mean)
    const size_t attnLds = (size_t)(16 * 1024 * 2 + 16 * 8 + 16) * sizeof(float);
    (void)hipFuncSetAttribute((const void*)attn_wmma,
                              hipFuncAttributeMaxDynamicSharedMemorySize,
                              (int)attnLds);
    attn_wmma<<<B_SZ * (N_SZ / 16), 128, attnLds, stream>>>(
        q_h, k_h, v_t, ctx, out + (size_t)B_SZ * N_SZ * QD);

    // 4) output projection -> fp32 first half of d_out
    gemm_wmma<2><<<gGemm, 128, 0, stream>>>(ctx, wo16, b_o, (void*)out, R, QD, QD);
}
